// Prototypical_80607946211607
// MI455X (gfx1250) — matrix operand order repaired and checked by reading, not compiled, not run on hardware
//
#include <hip/hip_runtime.h>

typedef __attribute__((ext_vector_type(16))) _Float16 v16h;
typedef __attribute__((ext_vector_type(8)))  _Float16 v8h;
typedef __attribute__((ext_vector_type(8)))  float    v8f;
typedef __attribute__((ext_vector_type(4)))  int      v4i;

#define NCLS 100
#define NSUP 10
#define NQRY 10
#define NIMG 2000
#define NSUPIMG 1000
#define BN_EPS 0.001f

#define GLOBAL_AS __attribute__((address_space(1)))
#define LDS_AS    __attribute__((address_space(3)))

#if defined(__AMDGCN__) && __has_builtin(__builtin_amdgcn_global_load_async_to_lds_b128) && \
    __has_builtin(__builtin_amdgcn_s_wait_asynccnt)
#define USE_ASYNC_LDS 1
#else
#define USE_ASYNC_LDS 0
#endif

// ---------------- param prep: fold BN into per-channel scale/bias ----------------
__global__ void prep_affine_kernel(const float* __restrict__ b, const float* __restrict__ gamma,
                                   const float* __restrict__ beta, const float* __restrict__ mean,
                                   const float* __restrict__ var,
                                   float* __restrict__ scale, float* __restrict__ bias) {
  int c = threadIdx.x;
  if (c < 64) {
    float s = gamma[c] * rsqrtf(var[c] + BN_EPS);
    scale[c] = s;
    bias[c]  = (b[c] - mean[c]) * s + beta[c];
  }
}

// Pack HWIO f32 (3,3,64,64) weights into WMMA B-fragment order (f16).
// B fragment (32x16, K x N) for k-chunk kt, n-tile nt:
//   lane L holds column N = nt*16 + (L&15), elements j=0..15 are K = kt*32 + (j>>3)*16 + (L>>4)*8 + (j&7).
// Flat K = (dy*3+dx)*64 + ci.
__global__ void pack_w_kernel(const float* __restrict__ w, _Float16* __restrict__ wp) {
  int e = blockIdx.x * blockDim.x + threadIdx.x;
  if (e >= 18 * 4 * 32 * 16) return;
  int j    = e & 15;
  int lane = (e >> 4) & 31;
  int tile = e >> 9;
  int nt = tile & 3;
  int kt = tile >> 2;
  int K = kt * 32 + ((j >> 3) << 4) + ((lane >> 4) << 3) + (j & 7);
  int N = nt * 16 + (lane & 15);
  int r  = K >> 6;   // dy*3+dx
  int ci = K & 63;
  wp[e] = (_Float16)w[(r * 64 + ci) * 64 + N];
}

// ---------------- layer 1: direct conv (cin=1, K=9 too small for WMMA) ----------------
__global__ void conv1_kernel(const float* __restrict__ support, const float* __restrict__ query,
                             const float* __restrict__ w1, const float* __restrict__ scale,
                             const float* __restrict__ bias, _Float16* __restrict__ x1) {
  __shared__ float in30[30 * 30];
  __shared__ float wl[576];
  __shared__ float sl[64], bl[64];
  const int n = blockIdx.x;
  const int tid = threadIdx.x;
  const float* src = (n < NSUPIMG) ? (support + (size_t)n * 784)
                                   : (query + (size_t)(n - NSUPIMG) * 784);
  for (int i = tid; i < 900; i += 256) in30[i] = 0.f;
  for (int i = tid; i < 576; i += 256) wl[i] = w1[i];
  if (tid < 64) { sl[tid] = scale[tid]; bl[tid] = bias[tid]; }
  __syncthreads();
  for (int i = tid; i < 784; i += 256) {
    int y = i / 28, x = i % 28;
    in30[(y + 1) * 30 + (x + 1)] = src[i];
  }
  __syncthreads();
  for (int o = tid; o < 14 * 14 * 64; o += 256) {
    int c = o & 63;
    int q = o >> 6;
    int px = q % 14, py = q / 14;
    float m = 0.f;  // ReLU outputs are >= 0
#pragma unroll
    for (int sy = 0; sy < 2; ++sy)
#pragma unroll
      for (int sx = 0; sx < 2; ++sx) {
        int y = 2 * py + sy, x = 2 * px + sx;
        float acc = 0.f;
#pragma unroll
        for (int dy = 0; dy < 3; ++dy)
#pragma unroll
          for (int dx = 0; dx < 3; ++dx)
            acc += in30[(y + dy) * 30 + (x + dx)] * wl[(dy * 3 + dx) * 64 + c];
        float v = fmaxf(acc * sl[c] + bl[c], 0.f);
        m = fmaxf(m, v);
      }
    x1[((size_t)n * 196 + q) * 64 + c] = (_Float16)m;
  }
}

// ---------------- layers 2-4: implicit-GEMM conv via v_wmma_f32_16x16x32_f16 ----------------
// One block (8 wave32s) per image. Activations staged in LDS with zero halo (SAME pad),
// via GLOBAL_LOAD_ASYNC_TO_LDS_B128 (ASYNCcnt) when available.
// M = spatial positions, N = 64 out channels. Each wave owns one 16-row M strip and all
// four N tiles: A fragment loaded once per K chunk (2 contiguous ds_load_b128), then
// 4 B fragments + 4 WMMAs into 4 accumulators (maximal A/B reuse per instruction).
// K = 576 as 18 chunks of 32; chunk kt has fixed (dy,dx)=decode(kt>>1) and channel
// range ci in [(kt&1)*32, +32), making per-lane A loads channel-contiguous and
// matching the ISA 16-bit A layout:
//   lane L row m=L&15; elements j=0..7 -> K_local = 8*(L>>4)+j ; j=8..15 -> 16+8*(L>>4)+(j-8).
template <int H, int W, bool LAST>
__global__ void conv_wmma_kernel(const _Float16* __restrict__ xin, const _Float16* __restrict__ wp,
                                 const float* __restrict__ scale, const float* __restrict__ bias,
                                 _Float16* __restrict__ xout, float* __restrict__ zout) {
  constexpr int HP = H + 2, WP = W + 2;
  constexpr int NPOS = H * W;
  constexpr int MT = (NPOS + 15) / 16;
  __shared__ alignas(32) _Float16 act[HP * WP * 64];
  __shared__ alignas(32) _Float16 outb[NPOS * 64];
  __shared__ float sl[64], bl[64];
  const int n = blockIdx.x;
  const int tid = threadIdx.x;
  const int wave = tid >> 5;
  const int lane = tid & 31;

  {  // zero halo buffer
    uint32_t* a32 = reinterpret_cast<uint32_t*>(act);
    for (int i = tid; i < HP * WP * 32; i += 256) a32[i] = 0u;
  }
  if (tid < 64) { sl[tid] = scale[tid]; bl[tid] = bias[tid]; }
  __syncthreads();
  // interior copy (source fully contiguous per image); rows are 128B-aligned in LDS
#if USE_ASYNC_LDS
  {
    const char* src = reinterpret_cast<const char*>(xin + (size_t)n * NPOS * 64);
    for (int i = tid; i < NPOS * 8; i += 256) {  // 16B chunks
      int y = i / (W * 8);
      int r = i - y * (W * 8);
      __builtin_amdgcn_global_load_async_to_lds_b128(
          (GLOBAL_AS v4i*)(void*)(src + (size_t)i * 16),
          (LDS_AS v4i*)(void*)&act[((y + 1) * WP + 1) * 64 + r * 8],
          0, 0);
    }
    __builtin_amdgcn_s_wait_asynccnt(0);
  }
#else
  {
    const uint32_t* s32 = reinterpret_cast<const uint32_t*>(xin + (size_t)n * NPOS * 64);
    uint32_t* a32 = reinterpret_cast<uint32_t*>(act);
    for (int i = tid; i < H * W * 32; i += 256) {
      int y = i / (W * 32);
      int r = i % (W * 32);
      a32[((y + 1) * WP + 1) * 32 + r] = s32[i];
    }
  }
#endif
  __syncthreads();

  const int g = lane >> 4;
  const int ma = lane & 15;

  for (int mt = wave; mt < MT; mt += 8) {
    int p = mt * 16 + ma;
    if (p >= NPOS) p = NPOS - 1;  // clamp address; those C rows are never stored
    const int y = p / W, x = p % W;
    v8f accs[4] = {};
#pragma unroll
    for (int kt = 0; kt < 18; ++kt) {
      const int ktr = kt >> 1;
      const int dy = ktr / 3, dx = ktr % 3;
      const int ciB = (kt & 1) * 32;
      const int base = ((y + dy) * WP + (x + dx)) * 64 + ciB + 8 * g;
      const v8h a0 = *reinterpret_cast<const v8h*>(&act[base]);
      const v8h a1 = *reinterpret_cast<const v8h*>(&act[base + 16]);
      v16h a;
#pragma unroll
      for (int i = 0; i < 8; ++i) { a[i] = a0[i]; a[i + 8] = a1[i]; }
#pragma unroll
      for (int nt = 0; nt < 4; ++nt) {
        const v16h b =
            *reinterpret_cast<const v16h*>(wp + ((size_t)(kt * 4 + nt) * 32 + lane) * 16);
        accs[nt] = __builtin_amdgcn_wmma_f32_16x16x32_f16(false, a, false, b, (short)0,
                                                          accs[nt], false, false);
      }
    }
    // C layout: vgpr r, lane L -> row M = 8*(L>>4)+r, col N = L&15
#pragma unroll
    for (int nt = 0; nt < 4; ++nt) {
      const int c = nt * 16 + (lane & 15);
      const float s = sl[c], bb = bl[c];
#pragma unroll
      for (int r = 0; r < 8; ++r) {
        int pr = mt * 16 + 8 * g + r;
        if (pr < NPOS) {
          float v = fmaxf(accs[nt][r] * s + bb, 0.f);
          outb[pr * 64 + c] = (_Float16)v;
        }
      }
    }
  }
  __syncthreads();

  // 2x2 VALID maxpool
  constexpr int HO = H / 2, WO = W / 2;
  for (int o = tid; o < HO * WO * 64; o += 256) {
    const int c = o & 63;
    const int q = o >> 6;
    const int px = q % WO, py = q / WO;
    float m0 = (float)outb[((2 * py) * W + 2 * px) * 64 + c];
    float m1 = (float)outb[((2 * py) * W + 2 * px + 1) * 64 + c];
    float m2 = (float)outb[((2 * py + 1) * W + 2 * px) * 64 + c];
    float m3 = (float)outb[((2 * py + 1) * W + 2 * px + 1) * 64 + c];
    float m = fmaxf(fmaxf(m0, m1), fmaxf(m2, m3));
    if (LAST) zout[(size_t)n * 64 + c] = m;
    else      xout[((size_t)n * HO * WO + q) * 64 + c] = (_Float16)m;
  }
}

// ---------------- prototypes, distances, loss, accuracy ----------------
__global__ void proto_kernel(const float* __restrict__ z, float* __restrict__ proto) {
  int cls = blockIdx.x, d = threadIdx.x;
  float s = 0.f;
  for (int i = 0; i < NSUP; ++i) s += z[((size_t)cls * NSUP + i) * 64 + d];
  proto[cls * 64 + d] = s * (1.f / NSUP);
}

__global__ void init_accum_kernel(float* accum) {
  if (threadIdx.x < 2) accum[threadIdx.x] = 0.f;
}

__global__ void loss_kernel(const float* __restrict__ z, const float* __restrict__ proto,
                            float* __restrict__ accum) {
  __shared__ float zq[64];
  __shared__ float dst[NCLS];
  const int q = blockIdx.x, tid = threadIdx.x;
  if (tid < 64) zq[tid] = z[(size_t)(NSUPIMG + q) * 64 + tid];
  __syncthreads();
  if (tid < NCLS) {
    float d = 0.f;
    const float* pr = proto + tid * 64;
    for (int k = 0; k < 64; ++k) { float t = zq[k] - pr[k]; d += t * t; }
    dst[tid] = d;
  }
  __syncthreads();
  if (tid == 0) {
    float dmin = dst[0]; int amin = 0;
    for (int c = 1; c < NCLS; ++c) { float d = dst[c]; if (d < dmin) { dmin = d; amin = c; } }
    float se = 0.f;
    for (int c = 0; c < NCLS; ++c) se += expf(dmin - dst[c]);
    const int label = q / NQRY;
    const float loss = dst[label] - dmin + logf(se);  // = d[label] + LSE(-d)
    atomicAdd(&accum[0], loss);
    atomicAdd(&accum[1], (amin == label) ? 1.f : 0.f);
  }
}

__global__ void finalize_kernel(const float* __restrict__ accum, float* __restrict__ out) {
  if (threadIdx.x == 0) {
    out[0] = accum[0] * (1.f / (NCLS * NQRY));
    out[1] = accum[1] * (1.f / (NCLS * NQRY));
  }
}

// ---------------- host side ----------------
extern "C" void kernel_launch(void* const* d_in, const int* in_sizes, int n_in,
                              void* d_out, int out_size, void* d_ws, size_t ws_size,
                              hipStream_t stream) {
  (void)in_sizes; (void)n_in; (void)out_size; (void)ws_size;
  const float* support = (const float*)d_in[0];
  const float* query   = (const float*)d_in[1];
  const float *pw[4], *pb[4], *pg[4], *pbe[4], *pm[4], *pv[4];
  for (int l = 0; l < 4; ++l) {
    pw[l]  = (const float*)d_in[2 + 6 * l + 0];
    pb[l]  = (const float*)d_in[2 + 6 * l + 1];
    pg[l]  = (const float*)d_in[2 + 6 * l + 2];
    pbe[l] = (const float*)d_in[2 + 6 * l + 3];
    pm[l]  = (const float*)d_in[2 + 6 * l + 4];
    pv[l]  = (const float*)d_in[2 + 6 * l + 5];
  }

  char* base = (char*)d_ws;
  size_t off = 0;
  auto alloc = [&](size_t bytes) -> void* {
    void* p = base + off;
    off = (off + bytes + 255) & ~((size_t)255);
    return p;
  };
  _Float16* x1 = (_Float16*)alloc((size_t)NIMG * 196 * 64 * 2);  // 14x14x64
  _Float16* x2 = (_Float16*)alloc((size_t)NIMG * 49 * 64 * 2);   // 7x7x64
  _Float16* x3 = (_Float16*)alloc((size_t)NIMG * 9 * 64 * 2);    // 3x3x64
  float* z     = (float*)alloc((size_t)NIMG * 64 * 4);
  float* proto = (float*)alloc((size_t)NCLS * 64 * 4);
  _Float16* wpk[3];
  for (int i = 0; i < 3; ++i) wpk[i] = (_Float16*)alloc((size_t)18 * 4 * 32 * 16 * 2);
  float* sb    = (float*)alloc((size_t)4 * 128 * 4);  // per layer: [scale(64) | bias(64)]
  float* accum = (float*)alloc(2 * 4);

  for (int l = 0; l < 4; ++l)
    prep_affine_kernel<<<1, 64, 0, stream>>>(pb[l], pg[l], pbe[l], pm[l], pv[l],
                                             sb + l * 128, sb + l * 128 + 64);
  for (int l = 0; l < 3; ++l)
    pack_w_kernel<<<144, 256, 0, stream>>>(pw[l + 1], wpk[l]);

  conv1_kernel<<<NIMG, 256, 0, stream>>>(support, query, pw[0], sb + 0, sb + 64, x1);
  conv_wmma_kernel<14, 14, false><<<NIMG, 256, 0, stream>>>(x1, wpk[0], sb + 128, sb + 192, x2, nullptr);
  conv_wmma_kernel<7, 7, false><<<NIMG, 256, 0, stream>>>(x2, wpk[1], sb + 256, sb + 320, x3, nullptr);
  conv_wmma_kernel<3, 3, true><<<NIMG, 256, 0, stream>>>(x3, wpk[2], sb + 384, sb + 448, nullptr, z);

  proto_kernel<<<NCLS, 64, 0, stream>>>(z, proto);
  init_accum_kernel<<<1, 32, 0, stream>>>(accum);
  loss_kernel<<<NCLS * NQRY, 128, 0, stream>>>(z, proto, accum);
  finalize_kernel<<<1, 32, 0, stream>>>(accum, (float*)d_out);
}
